// ChunkedCrossAttention_79869211836494
// MI455X (gfx1250) — matrix operand order repaired and checked by reading, not compiled, not run on hardware
//
#include <hip/hip_runtime.h>
#include <cstdint>

// ---------------------------------------------------------------------------
// ChunkedCrossAttention for MI455X (gfx1250), wave32 + WMMA bf16.
// Pipeline: rmsnorm -> Q/K/V GEMMs (bf16 WMMA) -> flash attention (WMMA) ->
//           O GEMM with shift+residual epilogue.
// All WMMA fragments are fed by contiguous 16B LDS loads (ds_load_b128):
//  - GEMM weight tiles staged transposed (Bt[n][k]) via column-per-thread
//    coalesced loads + vector ds_store_b128
//  - V stored transposed in global ([b,c,h,dk,j]) by the V-projection epilogue
// GEMM uses BK=64: 8 WMMAs per barrier pair.
// ---------------------------------------------------------------------------

typedef __attribute__((ext_vector_type(16))) __bf16 v16bf;
typedef __attribute__((ext_vector_type(8)))  __bf16 v8bf;
typedef __attribute__((ext_vector_type(8)))  float  v8f;

#define D_MODEL   1024
#define HDIM      1024   // N_HEADS * D_K
#define N_HEADS   16
#define D_KK      64
#define CHUNK_LEN 64
#define BB        4
#define CC        16
#define SSEQ      1024

static __device__ __forceinline__ v8f zero8() {
  v8f z = {0.f, 0.f, 0.f, 0.f, 0.f, 0.f, 0.f, 0.f};
  return z;
}

// Build a 16-element bf16 fragment from two 16B-aligned LDS chunks.
static __device__ __forceinline__ v16bf frag2(const __bf16* p0, const __bf16* p1) {
  v8bf lo = *reinterpret_cast<const v8bf*>(p0);
  v8bf hi = *reinterpret_cast<const v8bf*>(p1);
  return __builtin_shufflevector(lo, hi, 0, 1, 2, 3, 4, 5, 6, 7,
                                 8, 9, 10, 11, 12, 13, 14, 15);
}

// -------------------------------------------------------------------- rmsnorm
// hn[b, t, :] = rmsnorm(h[b, t+63, :]) * g   for t <= 960, else 0
__global__ __launch_bounds__(256) void rmsnorm_shift_kernel(
    const float* __restrict__ h, const float* __restrict__ g,
    float* __restrict__ hn) {
  const int row = blockIdx.x;            // b*1024 + t
  const int b = row >> 10, trow = row & 1023;
  float* out = hn + (size_t)row * D_MODEL;
  if (trow > SSEQ - CHUNK_LEN) {         // padded rows are exactly zero
    for (int i = threadIdx.x; i < D_MODEL; i += 256) out[i] = 0.f;
    return;
  }
  const float* x = h + ((size_t)b * SSEQ + trow + CHUNK_LEN - 1) * D_MODEL;
  __shared__ float red[256];
  float p = 0.f;
  for (int i = threadIdx.x; i < D_MODEL; i += 256) { float v = x[i]; p += v * v; }
  red[threadIdx.x] = p;
  __syncthreads();
  for (int s = 128; s > 0; s >>= 1) {
    if (threadIdx.x < s) red[threadIdx.x] += red[threadIdx.x + s];
    __syncthreads();
  }
  const float rms = rsqrtf(red[0] * (1.0f / D_MODEL) + 1e-8f);
  for (int i = threadIdx.x; i < D_MODEL; i += 256) out[i] = x[i] * g[i] * rms;
}

// ------------------------------------------------------------------ residual
__global__ __launch_bounds__(256) void init_out_kernel(
    const float* __restrict__ h, float* __restrict__ out, int n) {
  int i = blockIdx.x * 256 + threadIdx.x;
  if (i < n) out[i] = h[i];
}

// --------------------------------------------------------------- bf16 GEMM
// Out = A[M,1024] @ W[1024,1024] + bias. 128x64 block tile, 8 waves,
// each wave does a 16x64 strip (4 WMMA N-tiles), BK = 64 (2 k-steps/stage).
// mode 0: bf16 store permuted to [b, c, head, rin, dk]   (rcb = log2 rows/chunk)
// mode 3: bf16 store transposed  [b, c, head, dk, rin]   (V for attention)
// mode 2: fp32 shifted residual add into d_out
__global__ __launch_bounds__(256) void gemm_bf16_kernel(
    const float* __restrict__ A, const float* __restrict__ W,
    const float* __restrict__ bias, void* __restrict__ dst,
    int mode, int rcb) {
  __shared__ __align__(16) __bf16 As[128 * 64];   // row-major [m][k]
  __shared__ __align__(16) __bf16 Bt[64 * 64];    // TRANSPOSED [n][k]
  const int tid = threadIdx.x;
  const int wave = tid >> 5, lane = tid & 31;
  const int l15 = lane & 15;
  const int hi = (lane >= 16) ? 1 : 0;
  const int m0 = blockIdx.x * 128;
  const int n0 = blockIdx.y * 64;

  v8f acc[4];
#pragma unroll
  for (int i = 0; i < 4; ++i) acc[i] = zero8();

  const int Mr = wave * 16 + l15;
  const int kbA = hi ? 8 : 0;
  const int kbB = hi ? 16 : 0;

  for (int k0 = 0; k0 < D_MODEL; k0 += 64) {
    // stage A (fp32 -> bf16): 128x64, vectorized 16B LDS stores
#pragma unroll
    for (int ch = tid; ch < 1024; ch += 256) {
      const int row = ch >> 3, cb = (ch & 7) * 8;
      const float* src = A + (size_t)(m0 + row) * D_MODEL + k0 + cb;
      union { __bf16 e[8]; uint4 u; } t;
#pragma unroll
      for (int j = 0; j < 8; ++j) t.e[j] = (__bf16)src[j];
      *reinterpret_cast<uint4*>(&As[row * 64 + cb]) = t.u;
    }
    // stage W tile TRANSPOSED Bt[n][k]: thread owns one n-column, a 16-deep
    // k-strip; global loads coalesce across lanes, LDS stores are b128.
    {
      const int n = tid & 63;
      const int kg = (tid >> 6) * 16;
      const float* src = W + (size_t)(k0 + kg) * HDIM + n0 + n;
      union { __bf16 e[16]; uint4 u[2]; } t;
#pragma unroll
      for (int j = 0; j < 16; ++j) t.e[j] = (__bf16)src[(size_t)j * HDIM];
      uint4* d = reinterpret_cast<uint4*>(&Bt[n * 64 + kg]);
      d[0] = t.u[0];
      d[1] = t.u[1];
    }
    __syncthreads();

#pragma unroll
    for (int ks = 0; ks < 64; ks += 32) {
      const v16bf a = frag2(&As[Mr * 64 + ks + kbA],
                            &As[Mr * 64 + ks + 16 + kbA]);
#pragma unroll
      for (int nt = 0; nt < 4; ++nt) {
        const int nn = nt * 16 + l15;
        const v16bf bf = frag2(&Bt[nn * 64 + ks + kbA],
                               &Bt[nn * 64 + ks + 16 + kbA]);
        acc[nt] = __builtin_amdgcn_wmma_f32_16x16x32_bf16(
            false, a, false, bf, (short)0, acc[nt], false, false);
      }
    }
    __syncthreads();
  }

  // epilogue: C/D layout -> element (M = r + 8*hi, N = l15) per N-tile
#pragma unroll
  for (int nt = 0; nt < 4; ++nt) {
    const int gn = n0 + nt * 16 + l15;
    const float bv = bias[gn];
#pragma unroll
    for (int r = 0; r < 8; ++r) {
      const int gm = m0 + wave * 16 + r + (hi ? 8 : 0);
      const float val = acc[nt][r] + bv;
      if (mode == 2) {
        float* D = (float*)dst;
        const int b = gm >> 10, trow = gm & 1023;
        const int t = trow + (CHUNK_LEN - 1);
        if (t < SSEQ)
          D[((size_t)(b * SSEQ + t) << 10) + gn] += val;   // out pre-set to h
      } else {
        __bf16* D = (__bf16*)dst;
        const int b = gm >> (4 + rcb);
        const int c = (gm >> rcb) & 15;
        const int rin = gm & ((1 << rcb) - 1);
        const int hh = gn >> 6, dk = gn & 63;
        size_t idx;
        if (mode == 0)
          idx = ((size_t)(((((b * 16 + c) * 16 + hh) << rcb) + rin)) << 6) + dk;
        else  // mode 3: transposed V layout [b,c,h,dk,rin]
          idx = ((size_t)((((b * 16 + c) * 16 + hh) << 6) + dk) << rcb) + rin;
        D[idx] = (__bf16)val;
      }
    }
  }
}

// ------------------------------------------------------------- attention
// One block per (b, c, head): Q 64x64, K 256x64 (row-major j,d),
// V 64x256 (TRANSPOSED d,j) bf16. Flash-style online softmax over 4 panels
// of 64 keys. 4 waves; wave w owns query rows 16w..16w+15.
__global__ __launch_bounds__(128) void attn_kernel(
    const __bf16* __restrict__ q, const __bf16* __restrict__ k,
    const __bf16* __restrict__ vT, float* __restrict__ obuf) {
  __shared__ __align__(16) __bf16 Qs[64 * 64];   // [i][d]
  __shared__ __align__(16) __bf16 Ks[64 * 64];   // [j][d]
  __shared__ __align__(16) __bf16 Vt[64 * 64];   // [d][j]  (panel)
  __shared__ __align__(16) __bf16 Ps[64 * 64];   // [i][j]  4 waves x 16 x 64
  const int bh = blockIdx.x;                     // ((b*16+c)*16 + head)
  const int tid = threadIdx.x, wave = tid >> 5, lane = tid & 31;
  const int l15 = lane & 15;
  const int hi = (lane >= 16) ? 1 : 0;
  const float scale = 0.125f;                    // 1/sqrt(64)

  {  // load Q tile (8KB linear)
    const uint4* src = reinterpret_cast<const uint4*>(q + (size_t)bh * 64 * 64);
    uint4* d = reinterpret_cast<uint4*>(Qs);
    for (int i = tid; i < 512; i += 128) d[i] = src[i];
  }
  __syncthreads();

  // pre-gather Q A-fragments for the two depth-steps
  const int Mr = wave * 16 + l15;
  const int kbA = hi ? 8 : 0;
  const int kbB = hi ? 16 : 0;
  v16bf aQ[2];
#pragma unroll
  for (int db = 0; db < 2; ++db)
    aQ[db] = frag2(&Qs[Mr * 64 + db * 32 + kbA],
                   &Qs[Mr * 64 + db * 32 + 16 + kbA]);

  float m_run[8], l_run[8];
  v8f oacc[4];
#pragma unroll
  for (int r = 0; r < 8; ++r) { m_run[r] = -1e30f; l_run[r] = 0.f; }
#pragma unroll
  for (int dt = 0; dt < 4; ++dt) oacc[dt] = zero8();

  for (int p = 0; p < 4; ++p) {
    __syncthreads();
    {  // stage K panel [j][d] (linear) and V panel [d][j-slice] (row-sliced)
      const uint4* ks =
          reinterpret_cast<const uint4*>(k + (size_t)bh * 256 * 64 + p * 64 * 64);
      uint4* kd = reinterpret_cast<uint4*>(Ks);
      const __bf16* vsrc = vT + (size_t)bh * 64 * 256 + p * 64;
      for (int i = tid; i < 512; i += 128) {
        kd[i] = ks[i];
        const int d = i >> 3, jcb = (i & 7) * 8;
        *reinterpret_cast<uint4*>(&Vt[d * 64 + jcb]) =
            *reinterpret_cast<const uint4*>(&vsrc[d * 256 + jcb]);
      }
    }
    __syncthreads();

    // S = Q K^T for this panel: 4 N-tiles of 16 keys, 2 depth steps each
    v8f sacc[4];
#pragma unroll
    for (int nt = 0; nt < 4; ++nt) sacc[nt] = zero8();
#pragma unroll
    for (int nt = 0; nt < 4; ++nt) {
      const int jj = nt * 16 + l15;
#pragma unroll
      for (int db = 0; db < 2; ++db) {
        // B[k=d][n=j] = K[j][d]: contiguous along d in Ks row jj
        const v16bf bK = frag2(&Ks[jj * 64 + db * 32 + kbA],
                               &Ks[jj * 64 + db * 32 + 16 + kbA]);
        sacc[nt] = __builtin_amdgcn_wmma_f32_16x16x32_bf16(
            false, aQ[db], false, bK, (short)0, sacc[nt], false, false);
      }
    }

    // online softmax: row stats live per VGPR index r (row = 16w + r + 8*hi)
    float mnew[8], alpha[8];
#pragma unroll
    for (int r = 0; r < 8; ++r) {
      float mx = sacc[0][r];
#pragma unroll
      for (int nt = 1; nt < 4; ++nt) mx = fmaxf(mx, sacc[nt][r]);
      mx *= scale;
#pragma unroll
      for (int off = 1; off < 16; off <<= 1)
        mx = fmaxf(mx, __shfl_xor(mx, off, 32));
      mnew[r] = fmaxf(m_run[r], mx);
      alpha[r] = __expf(m_run[r] - mnew[r]);
      m_run[r] = mnew[r];
    }
    float psum[8];
#pragma unroll
    for (int r = 0; r < 8; ++r) psum[r] = 0.f;
#pragma unroll
    for (int nt = 0; nt < 4; ++nt) {
#pragma unroll
      for (int r = 0; r < 8; ++r) {
        const float ev = __expf(sacc[nt][r] * scale - mnew[r]);
        psum[r] += ev;
        Ps[(wave * 16 + r + (hi ? 8 : 0)) * 64 + nt * 16 + l15] = (__bf16)ev;
      }
    }
#pragma unroll
    for (int r = 0; r < 8; ++r) {
      float s = psum[r];
#pragma unroll
      for (int off = 1; off < 16; off <<= 1) s += __shfl_xor(s, off, 32);
      l_run[r] = l_run[r] * alpha[r] + s;
    }
#pragma unroll
    for (int dt = 0; dt < 4; ++dt)
#pragma unroll
      for (int r = 0; r < 8; ++r) oacc[dt][r] *= alpha[r];

    // Each wave re-reads only its own Ps region: DS ops in-order per wave,
    // just drain the DS counter (no block barrier needed).
    asm volatile("s_wait_dscnt 0" ::: "memory");

    // O += P V  (A = P 16x64, B = V 64x64 gathered from Vt[d][j])
#pragma unroll
    for (int db = 0; db < 2; ++db) {
      const v16bf aP = frag2(&Ps[Mr * 64 + db * 32 + kbA],
                             &Ps[Mr * 64 + db * 32 + 16 + kbA]);
#pragma unroll
      for (int dt = 0; dt < 4; ++dt) {
        // B[k=j][n=d] = Vt[d][j]: contiguous along j in Vt row d
        const int dd = dt * 16 + l15;
        const v16bf bV = frag2(&Vt[dd * 64 + db * 32 + kbA],
                               &Vt[dd * 64 + db * 32 + 16 + kbA]);
        oacc[dt] = __builtin_amdgcn_wmma_f32_16x16x32_bf16(
            false, aP, false, bV, (short)0, oacc[dt], false, false);
      }
    }
  }

  // normalize and write fp32 obuf[(b*16+c)*64 + i][head*64 + d]
  const int bc = bh >> 4, head = bh & 15;
#pragma unroll
  for (int dt = 0; dt < 4; ++dt)
#pragma unroll
    for (int r = 0; r < 8; ++r) {
      const int i = wave * 16 + r + (hi ? 8 : 0);
      const int row = bc * 64 + i;
      const int col = head * 64 + dt * 16 + l15;
      obuf[(size_t)row * 1024 + col] = oacc[dt][r] / l_run[r];
    }
}

// ------------------------------------------------------------------ launcher
extern "C" void kernel_launch(void* const* d_in, const int* in_sizes, int n_in,
                              void* d_out, int out_size, void* d_ws,
                              size_t ws_size, hipStream_t stream) {
  (void)in_sizes; (void)n_in; (void)out_size; (void)ws_size;
  const float* h  = (const float*)d_in[0];
  const float* e  = (const float*)d_in[1];
  const float* g  = (const float*)d_in[2];
  const float* Wq = (const float*)d_in[3];
  const float* bq = (const float*)d_in[4];
  const float* Wk = (const float*)d_in[5];
  const float* bk = (const float*)d_in[6];
  const float* Wv = (const float*)d_in[7];
  const float* bv = (const float*)d_in[8];
  const float* Wo = (const float*)d_in[9];
  const float* bo = (const float*)d_in[10];
  float* out = (float*)d_out;

  // workspace layout (bytes):
  //   hn   fp32 4096x1024 : 16777216
  //   q    bf16 4096x1024 :  8388608
  //   k    bf16 16384x1024: 33554432
  //   vT   bf16 16384x1024: 33554432   ([b,c,h,dk,j])
  //   obuf fp32 4096x1024 : 16777216   (total ~104 MB)
  char* ws = (char*)d_ws;
  float*  hn   = (float*)(ws);
  __bf16* qb   = (__bf16*)(ws + 16777216);
  __bf16* kbuf = (__bf16*)(ws + 25165824);
  __bf16* vbuf = (__bf16*)(ws + 58720256);
  float*  obuf = (float*)(ws + 92274688);

  rmsnorm_shift_kernel<<<4096, 256, 0, stream>>>(h, g, hn);
  init_out_kernel<<<(BB * SSEQ * D_MODEL + 255) / 256, 256, 0, stream>>>(
      h, out, BB * SSEQ * D_MODEL);

  gemm_bf16_kernel<<<dim3(32, 16), 256, 0, stream>>>(hn, Wq, bq, (void*)qb, 0, 6);
  gemm_bf16_kernel<<<dim3(128, 16), 256, 0, stream>>>(e, Wk, bk, (void*)kbuf, 0, 8);
  gemm_bf16_kernel<<<dim3(128, 16), 256, 0, stream>>>(e, Wv, bv, (void*)vbuf, 3, 8);

  attn_kernel<<<BB * CC * N_HEADS, 128, 0, stream>>>(qb, kbuf, vbuf, obuf);

  gemm_bf16_kernel<<<dim3(32, 16), 256, 0, stream>>>(obuf, Wo, bo, (void*)out, 2, 0);
}
